// SemiPermeableAttention_41351945126019
// MI455X (gfx1250) — matrix operand order repaired and checked by reading, not compiled, not run on hardware
//
#include <hip/hip_runtime.h>
#include <hip/hip_bf16.h>

// ---------------- CDNA5 (gfx1250) types ----------------
typedef __bf16 bf16;
typedef __attribute__((ext_vector_type(16))) __bf16 v16bf;
typedef __attribute__((ext_vector_type(8)))  __bf16 v8bf;
typedef __attribute__((ext_vector_type(4)))  __bf16 v4bf;
typedef __attribute__((ext_vector_type(8)))  float  v8f;
typedef __attribute__((ext_vector_type(4)))  float  v4f;
typedef __attribute__((ext_vector_type(4)))  unsigned int v4u;
typedef __attribute__((ext_vector_type(8)))  int    v8i;
typedef __attribute__((ext_vector_type(4)))  int    v4i;

#define Dm   1024
#define Nn   2048
#define Bb   4
#define Hh   16
#define HD   64
#define BN   (Bb * Nn)                          // 8192 rows total
#define MAT_ELEMS ((size_t)Bb * Nn * Dm)        // 8388608
#define W_ELEMS   ((size_t)Dm * Dm)             // 1048576

static __device__ __forceinline__ v8f wmma_bf16(v16bf a, v16bf b, v8f c) {
  // D(16x16,f32) = A(16x32,bf16) * B(32x16,bf16) + C
  return __builtin_amdgcn_wmma_f32_16x16x32_bf16(
      /*neg_a=*/false, a, /*neg_b=*/false, b,
      /*c_mod=*/(short)0, c, /*reuse_a=*/false, /*reuse_b=*/false);
}

// ------------------------------------------------------------
// TDM: async-copy a 64(rows) x 32(k) bf16 tile of W into LDS.
// 2-D tensor descriptor (D#): data_size=2B, tensor dims 1024x1024,
// dim0 stride 1024, tile 32x64. lds_off is a byte offset into the
// workgroup LDS (wtile is the kernel's only LDS object, offset 0).
// ------------------------------------------------------------
static __device__ __forceinline__ void tdm_load_w_tile(const bf16* gptr,
                                                       unsigned lds_off) {
  unsigned long long ga = (unsigned long long)gptr;
  v4u g0;
  g0[0] = 1u;                                          // count=1 (valid), user mode
  g0[1] = lds_off;                                     // lds_addr [63:32]
  g0[2] = (unsigned)(ga & 0xFFFFFFFFu);                // global_addr [95:64]
  g0[3] = (unsigned)((ga >> 32) & 0x1FFFFFFu)          // global_addr [120:96]
          | 0x80000000u;                               // type=2 ("image") [127:126]
  v8i g1;
  g1[0] = 0x10000;                                     // data_size=1 (2 bytes)
  g1[1] = (int)(1024u << 16);                          // tensor_dim0 lo16 @ [63:48]
  g1[2] = (int)(1024u << 16);                          // dim0 hi=0 | tensor_dim1 lo16
  g1[3] = (int)(32u << 16);                            // dim1 hi=0 | tile_dim0=32
  g1[4] = 64;                                          // tile_dim1=64 (tile_dim2=0)
  g1[5] = 1024;                                        // tensor_dim0_stride lo32
  g1[6] = 0;                                           // stride hi | dim1_stride lo
  g1[7] = 0;
  v4i z4 = {0, 0, 0, 0};
#if __clang_major__ >= 23
  v8i z8 = {0, 0, 0, 0, 0, 0, 0, 0};
  __builtin_amdgcn_tensor_load_to_lds(g0, g1, z4, z4, z8, 0);
#else
  __builtin_amdgcn_tensor_load_to_lds(g0, g1, z4, z4, 0);
#endif
}

// ============================================================
// Kernel 0: bulk f32 -> bf16 conversion (native hardware cvt path)
// ============================================================
__global__ __launch_bounds__(256) void f32_to_bf16_kernel(
    const float* __restrict__ in, bf16* __restrict__ out, int n4) {
  int i = blockIdx.x * 256 + threadIdx.x;
  if (i >= n4) return;
  v4f x = ((const v4f*)in)[i];
  v4bf y;
#pragma unroll
  for (int j = 0; j < 4; ++j) y[j] = (bf16)x[j];
  ((v4bf*)out)[i] = y;
}

// ============================================================
// GEMM: Y[m,n] = sum_k X[m,k] * W[n,k] + bias[n]   (x @ W^T + b)
// 64x64 per wave (4x4 tiles), 8 waves / block. A-fragments via
// global b128 loads; W tiles via double-buffered TDM into LDS.
// TR_OUT: store output transposed as [B, D, N] (used for V).
// ============================================================
template <typename OutT, bool TR_OUT>
__global__ __launch_bounds__(256) void gemm_xWT_kernel(
    const bf16* __restrict__ X, const bf16* __restrict__ W,
    const float* __restrict__ bias, OutT* __restrict__ out) {
  __shared__ bf16 wtile[8][2][64 * 32];       // 8 waves x 2 buffers x 4KB

  const int lane = threadIdx.x & 31;
  const int wave = threadIdx.x >> 5;
  const int tile = blockIdx.x * 8 + wave;     // 0 .. (BN/64)*(D/64)-1 = 2047
  const int mtb = tile >> 4;                  // / (D/64)
  const int ntb = tile & 15;
  const int m0 = mtb * 64;
  const int n0 = ntb * 64;
  const int n = lane & 15;
  const int h = lane >> 4;

  const bf16* arow[4];
#pragma unroll
  for (int s = 0; s < 4; ++s) arow[s] = X + (size_t)(m0 + s * 16 + n) * Dm;
  const bf16* wbase = W + (size_t)n0 * Dm;
  const unsigned lds_off[2] = { (unsigned)(wave * 2 + 0) * 4096u,
                                (unsigned)(wave * 2 + 1) * 4096u };

  v8f acc[4][4] = {};

  // prime the pipeline: tile for k0 = 0 into buffer 0
  tdm_load_w_tile(wbase, lds_off[0]);

  for (int k0 = 0; k0 < Dm; k0 += 32) {
    const int cur = (k0 >> 5) & 1;
    if (k0 + 32 < Dm) {
      tdm_load_w_tile(wbase + k0 + 32, lds_off[cur ^ 1]);
      __builtin_amdgcn_s_wait_tensorcnt(1);   // current tile resident
    } else {
      __builtin_amdgcn_s_wait_tensorcnt(0);
    }

    __builtin_prefetch(arow[0] + k0 + 256, 0, 0);
    v16bf af[4], bf[4];
#pragma unroll
    for (int s = 0; s < 4; ++s) {
      v8bf lo = *(const v8bf*)(arow[s] + k0 + 8 * h);
      v8bf hi = *(const v8bf*)(arow[s] + k0 + 16 + 8 * h);
#pragma unroll
      for (int i = 0; i < 8; ++i) { af[s][i] = lo[i]; af[s][8 + i] = hi[i]; }
    }
    const bf16* wt = &wtile[wave][cur][0];
#pragma unroll
    for (int s = 0; s < 4; ++s) {
      v8bf lo = *(const v8bf*)(wt + (s * 16 + n) * 32 + 8 * h);
      v8bf hi = *(const v8bf*)(wt + (s * 16 + n) * 32 + 16 + 8 * h);
#pragma unroll
      for (int i = 0; i < 8; ++i) { bf[s][i] = lo[i]; bf[s][8 + i] = hi[i]; }
    }
#pragma unroll
    for (int ms = 0; ms < 4; ++ms)
#pragma unroll
      for (int ns = 0; ns < 4; ++ns)
        acc[ms][ns] = wmma_bf16(af[ms], bf[ns], acc[ms][ns]);
  }

#pragma unroll
  for (int ns = 0; ns < 4; ++ns) {
    const int col = n0 + ns * 16 + n;
    const float bv = bias[col];
#pragma unroll
    for (int ms = 0; ms < 4; ++ms) {
#pragma unroll
      for (int r = 0; r < 8; ++r) {
        const int row = m0 + ms * 16 + r + 8 * h;   // global row in [0, B*N)
        const float v = acc[ms][ns][r] + bv;
        size_t idx;
        if constexpr (TR_OUT) {
          // [B, D, N]: b = row / N, seq = row % N
          idx = (size_t)(row >> 11) * ((size_t)Dm * Nn) + (size_t)col * Nn + (row & (Nn - 1));
        } else {
          idx = (size_t)row * Dm + col;
        }
        if constexpr (__is_same(OutT, bf16))
          out[idx] = (bf16)v;
        else
          out[idx] = v;
      }
    }
  }
}

// ============================================================
// Causal flash attention: one wave per (b, head, 16-row q tile).
// Q,K bf16 [B, N, D]; V bf16 TRANSPOSED [B, D, N]. Online softmax f32.
// k0 <= q0+15 with k0 % 32 == 0 implies max key index = 2047: no clamps.
// ============================================================
__global__ __launch_bounds__(32) void attn_kernel(
    const bf16* __restrict__ Q, const bf16* __restrict__ K,
    const bf16* __restrict__ Vt, bf16* __restrict__ O) {
  __shared__ bf16 plds[16 * 32];   // P tile (16 q-rows x 32 keys)

  const int lane = threadIdx.x & 31;
  const int n = lane & 15;
  const int h = lane >> 4;
  const int q0  = blockIdx.x * 16;
  const int hd0 = blockIdx.y * HD;
  const size_t base = (size_t)blockIdx.z * Nn * Dm;   // same for [B,N,D] and [B,D,N]

  // Q A-fragments for d0 = 0 and 32 (reused across all key tiles)
  v16bf qa[2];
  {
    const bf16* qrow = Q + base + (size_t)(q0 + n) * Dm + hd0;
#pragma unroll
    for (int t = 0; t < 2; ++t) {
      v8bf lo = *(const v8bf*)(qrow + t * 32 + 16 * h);
      v8bf hi = *(const v8bf*)(qrow + t * 32 + 16 * h + 8);
#pragma unroll
      for (int i = 0; i < 8; ++i) { qa[t][i] = lo[i]; qa[t][8 + i] = hi[i]; }
    }
  }

  float mrun[8], lrun[8];
  v8f oacc[4] = {};
#pragma unroll
  for (int r = 0; r < 8; ++r) { mrun[r] = -1e30f; lrun[r] = 0.0f; }

  for (int k0 = 0; k0 <= q0 + 15; k0 += 32) {
    // ---- S = Q * K^T for 32 keys (two 16-wide WMMA column tiles) ----
    v8f s0acc = {}, s1acc = {};
    const bf16* krow0 = K + base + (size_t)(k0 + n) * Dm + hd0;
    const bf16* krow1 = K + base + (size_t)(k0 + 16 + n) * Dm + hd0;
#pragma unroll
    for (int t = 0; t < 2; ++t) {
      v16bf b0 = *(const v16bf*)(krow0 + t * 32 + 16 * h);
      v16bf b1 = *(const v16bf*)(krow1 + t * 32 + 16 * h);
      s0acc = wmma_bf16(qa[t], b0, s0acc);
      s1acc = wmma_bf16(qa[t], b1, s1acc);
    }

    // ---- online softmax (rows live across the 16 lanes of a half-wave) ----
#pragma unroll
    for (int r = 0; r < 8; ++r) {
      const int q = q0 + r + 8 * h;
      float s0 = s0acc[r] * 0.125f;            // 1/sqrt(64)
      float s1 = s1acc[r] * 0.125f;
      if (k0 + n > q)      s0 = -1e30f;        // causal mask
      if (k0 + 16 + n > q) s1 = -1e30f;
      float rmax = fmaxf(s0, s1);
#pragma unroll
      for (int mm = 1; mm < 16; mm <<= 1) rmax = fmaxf(rmax, __shfl_xor(rmax, mm, 32));
      const float mnew = fmaxf(mrun[r], rmax);
      const float corr = __expf(mrun[r] - mnew);
      const float p0 = __expf(s0 - mnew);
      const float p1 = __expf(s1 - mnew);
      float rsum = p0 + p1;
#pragma unroll
      for (int mm = 1; mm < 16; mm <<= 1) rsum += __shfl_xor(rsum, mm, 32);
      lrun[r] = lrun[r] * corr + rsum;
      mrun[r] = mnew;
#pragma unroll
      for (int ds = 0; ds < 4; ++ds) oacc[ds][r] *= corr;
      plds[(r + 8 * h) * 32 + n]      = (bf16)p0;
      plds[(r + 8 * h) * 32 + 16 + n] = (bf16)p1;
    }

    // ---- repack P from D-layout (LDS) into A-layout ----
    v16bf pa;
    {
      v8bf lo = *(const v8bf*)(&plds[n * 32 + 16 * h]);
      v8bf hi = *(const v8bf*)(&plds[n * 32 + 16 * h + 8]);
#pragma unroll
      for (int i = 0; i < 8; ++i) { pa[i] = lo[i]; pa[8 + i] = hi[i]; }
    }

    // ---- O += P * V: V^T gives contiguous 16-key vector loads ----
#pragma unroll
    for (int ds = 0; ds < 4; ++ds) {
      const v16bf vb = *(const v16bf*)(Vt + base +
                        (size_t)(hd0 + ds * 16 + n) * Nn + k0 + 16 * h);
      oacc[ds] = wmma_bf16(pa, vb, oacc[ds]);
    }
  }

  // ---- normalize and write bf16 output slice ----
#pragma unroll
  for (int r = 0; r < 8; ++r) {
    const float inv = 1.0f / lrun[r];
    const int q = q0 + r + 8 * h;
#pragma unroll
    for (int ds = 0; ds < 4; ++ds) {
      O[base + (size_t)q * Dm + hd0 + ds * 16 + n] = (bf16)(oacc[ds][r] * inv);
    }
  }
}

// ============================================================
extern "C" void kernel_launch(void* const* d_in, const int* in_sizes, int n_in,
                              void* d_out, int out_size, void* d_ws, size_t ws_size,
                              hipStream_t stream) {
  (void)in_sizes; (void)n_in; (void)out_size; (void)ws_size;
  const float* x_q  = (const float*)d_in[0];
  const float* x_kv = (const float*)d_in[1];
  const float* Wq = (const float*)d_in[2];
  const float* bq = (const float*)d_in[3];
  const float* Wk = (const float*)d_in[4];
  const float* bk = (const float*)d_in[5];
  const float* Wv = (const float*)d_in[6];
  const float* bv = (const float*)d_in[7];
  const float* Wo = (const float*)d_in[8];
  const float* bo = (const float*)d_in[9];

  // ---- workspace layout (bf16), total ~104 MB ----
  bf16* xq_bf  = (bf16*)d_ws;
  bf16* xkv_bf = xq_bf  + MAT_ELEMS;
  bf16* wq_bf  = xkv_bf + MAT_ELEMS;
  bf16* wk_bf  = wq_bf  + W_ELEMS;
  bf16* wv_bf  = wk_bf  + W_ELEMS;
  bf16* wo_bf  = wv_bf  + W_ELEMS;
  bf16* Qws    = wo_bf  + W_ELEMS;
  bf16* Kws    = Qws    + MAT_ELEMS;
  bf16* Vtws   = Kws    + MAT_ELEMS;   // transposed [B, D, N]
  bf16* AOws   = Vtws   + MAT_ELEMS;

  // ---- stage inputs/weights as bf16 (single HBM sweep) ----
  const int x4 = (int)(MAT_ELEMS / 4), w4 = (int)(W_ELEMS / 4);
  f32_to_bf16_kernel<<<(x4 + 255) / 256, 256, 0, stream>>>(x_q,  xq_bf,  x4);
  f32_to_bf16_kernel<<<(x4 + 255) / 256, 256, 0, stream>>>(x_kv, xkv_bf, x4);
  f32_to_bf16_kernel<<<(w4 + 255) / 256, 256, 0, stream>>>(Wq, wq_bf, w4);
  f32_to_bf16_kernel<<<(w4 + 255) / 256, 256, 0, stream>>>(Wk, wk_bf, w4);
  f32_to_bf16_kernel<<<(w4 + 255) / 256, 256, 0, stream>>>(Wv, wv_bf, w4);
  f32_to_bf16_kernel<<<(w4 + 255) / 256, 256, 0, stream>>>(Wo, wo_bf, w4);

  // ---- projections: 64x64 per wave, 8 waves per block, TDM W tiles ----
  const int gemm_blocks = (BN / 64) * (Dm / 64) / 8;   // 256
  gemm_xWT_kernel<bf16,  false><<<gemm_blocks, 256, 0, stream>>>(xq_bf,  wq_bf, bq, Qws);
  gemm_xWT_kernel<bf16,  false><<<gemm_blocks, 256, 0, stream>>>(xkv_bf, wk_bf, bk, Kws);
  gemm_xWT_kernel<bf16,  true ><<<gemm_blocks, 256, 0, stream>>>(xkv_bf, wv_bf, bv, Vtws);

  // ---- attention ----
  attn_kernel<<<dim3(Nn / 16, Hh, Bb), 32, 0, stream>>>(Qws, Kws, Vtws, AOws);

  // ---- output projection (f32 out) ----
  gemm_xWT_kernel<float, false><<<gemm_blocks, 256, 0, stream>>>(AOws, wo_bf, bo, (float*)d_out);
}
